// Custom_MHA_46420006535668
// MI455X (gfx1250) — matrix operand order repaired and checked by reading, not compiled, not run on hardware
//
#include <hip/hip_runtime.h>
#include <hip/hip_bf16.h>

// ---------------------------------------------------------------------------
// MHA forward for MI455X (gfx1250, wave32, WMMA + TDM).
//   B=2, S=2048, H=16, DK=64, D=1024
//   1) proj_qkv (x3): X_f32 @ W_f32 + b -> head-split f16; Q scaled by 1/8
//   2) attn: flash attention, causal; K tiles staged via tensor_load_to_lds
//   3) proj_out: O_f16 @ W_o + b_o -> f32 d_out
// ---------------------------------------------------------------------------

#define S_LEN    2048
#define N_HEADS  16
#define HEAD_DIM 64
#define D_MODEL  1024
#define BATCH    2

typedef __attribute__((ext_vector_type(16))) _Float16 v16h;
typedef __attribute__((ext_vector_type(8)))  _Float16 v8h;
typedef __attribute__((ext_vector_type(8)))  float    v8f;
typedef __attribute__((ext_vector_type(4)))  float    v4f;
typedef __attribute__((ext_vector_type(4)))  unsigned int v4u;
typedef __attribute__((ext_vector_type(8)))  int      v8i;
typedef __attribute__((ext_vector_type(4)))  int      v4i;

union V16H { v16h v; struct { v8h lo, hi; } c; };

__device__ __forceinline__ v16h frag16(const _Float16* p0, const _Float16* p1) {
  V16H u;
  u.c.lo = *(const v8h*)p0;   // 16B -> ds_load_b128 / global_load_b128
  u.c.hi = *(const v8h*)p1;
  return u.v;
}

__device__ __forceinline__ v8f wmma_ff16(v16h a, v16h b, v8f c) {
  return __builtin_amdgcn_wmma_f32_16x16x32_f16(false, a, false, b, (short)0, c,
                                                false, false);
}

__device__ __forceinline__ v8f zero8() {
  v8f z = {0.f, 0.f, 0.f, 0.f, 0.f, 0.f, 0.f, 0.f};
  return z;
}

// ---------------------------------------------------------------------------
// Kernel 1: projection GEMM, f32 in, f16 head-split out.
// Block tile 128(M) x 64(N), 8 waves x (16x64). K staged 64 at a time.
// LDS strides padded to 72 halves (144B = 36 banks) -> conflict-free b128.
// ---------------------------------------------------------------------------
__global__ __launch_bounds__(256)
void proj_qkv_kernel(const float* __restrict__ X, const float* __restrict__ W,
                     const float* __restrict__ bias, _Float16* __restrict__ Out,
                     float scale) {
  __shared__ _Float16 ldsA[128 * 72];  // [row][k]
  __shared__ _Float16 ldsB[64 * 72];   // [n][k]

  const int t    = threadIdx.x;
  const int lane = t & 31;
  const int wid  = t >> 5;
  const int lrow = lane & 15;
  const int hi   = lane >> 4;
  const int m0   = blockIdx.x * 128;
  const int n0   = blockIdx.y * 64;
  const int kb   = hi * 8;

  v8f acc[4];
#pragma unroll
  for (int nt = 0; nt < 4; ++nt) acc[nt] = zero8();

  for (int k0 = 0; k0 < D_MODEL; k0 += 64) {
    {  // stage A: 128 rows x 64 k, f32 -> f16 (32 elems / thread)
      const int row = t >> 1;
      const int c0  = (t & 1) * 32;
      const float* src = X + (size_t)(m0 + row) * D_MODEL + k0 + c0;
      _Float16* dst = &ldsA[row * 72 + c0];
#pragma unroll
      for (int i = 0; i < 32; i += 4) {
        v4f f = *(const v4f*)(src + i);
        dst[i + 0] = (_Float16)f.x;
        dst[i + 1] = (_Float16)f.y;
        dst[i + 2] = (_Float16)f.z;
        dst[i + 3] = (_Float16)f.w;
      }
    }
    {  // stage B: 64 k x 64 n, transposed to [n][k], f32 -> f16
      const int n   = t & 63;
      const int kbb = (t >> 6) * 16;
#pragma unroll
      for (int i = 0; i < 16; ++i)
        ldsB[n * 72 + kbb + i] =
            (_Float16)W[(size_t)(k0 + kbb + i) * D_MODEL + n0 + n];
    }
    __syncthreads();

    const int arow = wid * 16 + lrow;
#pragma unroll
    for (int kt = 0; kt < 2; ++kt) {
      const v16h af = frag16(&ldsA[arow * 72 + kt * 32 + kb],
                             &ldsA[arow * 72 + kt * 32 + 16 + kb]);
      v16h bf[4];
#pragma unroll
      for (int nt = 0; nt < 4; ++nt) {
        const int col = nt * 16 + lrow;
        bf[nt] = frag16(&ldsB[col * 72 + kt * 32 + kb],
                        &ldsB[col * 72 + kt * 32 + 16 + kb]);
      }
#pragma unroll
      for (int nt = 0; nt < 4; ++nt) acc[nt] = wmma_ff16(af, bf[nt], acc[nt]);
    }
    __syncthreads();
  }

  // Epilogue: bias, scale, scatter to head-split f16 layout (B,H,S,DK)
#pragma unroll
  for (int nt = 0; nt < 4; ++nt) {
    const int col = n0 + nt * 16 + lrow;
    const float bv = bias[col];
    const int h  = col >> 6;
    const int dk = col & 63;
#pragma unroll
    for (int r = 0; r < 8; ++r) {
      const int row = m0 + wid * 16 + r + 8 * hi;
      const int bb  = row >> 11;
      const int s   = row & 2047;
      const float val = (acc[nt][r] + bv) * scale;
      Out[(size_t)((bb * N_HEADS + h) * S_LEN + s) * HEAD_DIM + dk] =
          (_Float16)val;
    }
  }
}

// ---------------------------------------------------------------------------
// Kernel 2: flash attention, causal. 128 query rows / block (8 waves x 16).
// K tile (64x64 f16, row-major) staged by the Tensor Data Mover with LDS row
// padding (pad_interval = 32 DWORDs -> code 4, pad_amount = 4 DWORDs -> code 3)
// so rows land at stride 72 halves. V tile transposed manually.
// ---------------------------------------------------------------------------
__global__ __launch_bounds__(256)
void attn_kernel(const _Float16* __restrict__ Qh, const _Float16* __restrict__ Kh,
                 const _Float16* __restrict__ Vh, _Float16* __restrict__ Oc) {
  __shared__ _Float16 ldsK[64 * 72];      // [key][dk]
  __shared__ _Float16 ldsV[64 * 72];      // [dk][key]
  __shared__ _Float16 ldsP[8 * 16 * 72];  // per-wave P bounce

  const int t    = threadIdx.x;
  const int lane = t & 31;
  const int wid  = t >> 5;
  const int lrow = lane & 15;
  const int hi   = lane >> 4;
  const int kb   = hi * 8;
  const int q0   = blockIdx.x * 128;
  const int bh   = blockIdx.y;
  const int bb   = bh >> 4;
  const int h    = bh & 15;

  const _Float16* qb  = Qh + (size_t)bh * S_LEN * HEAD_DIM;
  const _Float16* kbp = Kh + (size_t)bh * S_LEN * HEAD_DIM;
  const _Float16* vbp = Vh + (size_t)bh * S_LEN * HEAD_DIM;

  const int qrow = q0 + wid * 16 + lrow;
  v16h qf[2];
#pragma unroll
  for (int kt = 0; kt < 2; ++kt) {
    const _Float16* p = qb + (size_t)qrow * HEAD_DIM + kt * 32;
    qf[kt] = frag16(p + kb, p + 16 + kb);
  }

  float m_r[8], l_r[8];
  v8f o[4];
#pragma unroll
  for (int r = 0; r < 8; ++r) { m_r[r] = -1.0e30f; l_r[r] = 0.f; }
#pragma unroll
  for (int dt = 0; dt < 4; ++dt) o[dt] = zero8();

  const int nj = q0 / 64 + 2;
  for (int jb = 0; jb < nj; ++jb) {
    const int kcol0 = jb * 64;

    if (wid == 0) {
      // TDM: DMA the 64x64 f16 K tile (row-major) into ldsK with row padding.
      // D# group0: count=1 | lds_addr | global_addr(57b) | type=2
      const unsigned long long ga =
          (unsigned long long)(uintptr_t)(kbp + (size_t)kcol0 * HEAD_DIM);
      const unsigned int ldsOff = (unsigned int)(uintptr_t)(&ldsK[0]);
      v4u g0 = {1u, ldsOff, (unsigned int)ga,
                (unsigned int)((ga >> 32) & 0x01FFFFFFu) | 0x80000000u};
      // D# group1:
      //  d0: data_size=1(2B)<<16 | pad_enable<<20 | pad_interval(4)<<22
      //      | pad_amount(3)<<25
      //  d1: tensor_dim0(64) << 16        (bits 79:48)
      //  d2: tensor_dim1(2048) << 16      (bits 111:80)
      //  d3: tile_dim0(64) << 16          (bits 127:112)
      //  d4: tile_dim1(64)                (bits 143:128)
      //  d5: tensor_dim0_stride(64)       (bits 207:160)
      v8i g1 = {(int)((1u << 16) | (1u << 20) | (4u << 22) | (3u << 25)),
                (int)(64u << 16), (int)(2048u << 16), (int)(64u << 16),
                64, 64, 0, 0};
      v4i gz4 = {0, 0, 0, 0};
      v8i gz8 = {0, 0, 0, 0, 0, 0, 0, 0};
      // 6-arg form on this toolchain: (g0, g1, g2, g3, g4, cpol)
      __builtin_amdgcn_tensor_load_to_lds(g0, g1, gz4, gz4, gz8, 0);
    }

    {  // stage V tile transposed: ldsV[dk][key]
      const int key = t >> 2;
      const int d0  = (t & 3) * 16;
      const _Float16* src = vbp + (size_t)(kcol0 + key) * HEAD_DIM + d0;
#pragma unroll
      for (int i = 0; i < 16; ++i) ldsV[(d0 + i) * 72 + key] = src[i];
    }
    if (jb + 1 < nj) {  // prefetch next V tile -> global_prefetch_b8
      __builtin_prefetch(vbp + (size_t)(kcol0 + 64 + (t >> 2)) * HEAD_DIM, 0, 1);
    }
    if (wid == 0) __builtin_amdgcn_s_wait_tensorcnt(0);
    __syncthreads();

    // scores S = Q @ K^T (16 x 64), f32 accum
    v8f s[4];
#pragma unroll
    for (int nt = 0; nt < 4; ++nt) s[nt] = zero8();
#pragma unroll
    for (int kt = 0; kt < 2; ++kt) {
      v16h bf[4];
#pragma unroll
      for (int nt = 0; nt < 4; ++nt) {
        const int col = nt * 16 + lrow;
        bf[nt] = frag16(&ldsK[col * 72 + kt * 32 + kb],
                        &ldsK[col * 72 + kt * 32 + 16 + kb]);
      }
#pragma unroll
      for (int nt = 0; nt < 4; ++nt) s[nt] = wmma_ff16(qf[kt], bf[nt], s[nt]);
    }

    // causal mask + online softmax
#pragma unroll
    for (int r = 0; r < 8; ++r) {
      const int rowg = q0 + wid * 16 + r + 8 * hi;
      float mx = -3.0e38f;
#pragma unroll
      for (int nt = 0; nt < 4; ++nt) {
        const int colg = kcol0 + nt * 16 + lrow;
        float sv = s[nt][r];
        if (colg > rowg) sv += -1.0e9f;
        s[nt][r] = sv;
        mx = fmaxf(mx, sv);
      }
#pragma unroll
      for (int d = 1; d < 16; d <<= 1) mx = fmaxf(mx, __shfl_xor(mx, d, 32));
      const float mn = fmaxf(m_r[r], mx);
      const float al = __expf(m_r[r] - mn);
      float rs = 0.f;
#pragma unroll
      for (int nt = 0; nt < 4; ++nt) {
        const float p = __expf(s[nt][r] - mn);
        s[nt][r] = p;
        rs += p;
      }
#pragma unroll
      for (int d = 1; d < 16; d <<= 1) rs += __shfl_xor(rs, d, 32);
      l_r[r] = l_r[r] * al + rs;
      m_r[r] = mn;
#pragma unroll
      for (int dt = 0; dt < 4; ++dt) o[dt][r] *= al;
    }

    // bounce P through per-wave LDS region (D-frag -> A-frag reshape)
    _Float16* pw = &ldsP[wid * 16 * 72];
#pragma unroll
    for (int nt = 0; nt < 4; ++nt)
#pragma unroll
      for (int r = 0; r < 8; ++r)
        pw[(r + 8 * hi) * 72 + nt * 16 + lrow] = (_Float16)s[nt][r];

    // O += P @ V
#pragma unroll
    for (int kt = 0; kt < 2; ++kt) {
      const _Float16* pp = pw + lrow * 72 + kt * 32;
      const v16h pf = frag16(pp + kb, pp + 16 + kb);
      v16h vf[4];
#pragma unroll
      for (int dt = 0; dt < 4; ++dt) {
        const int dk = dt * 16 + lrow;
        vf[dt] = frag16(&ldsV[dk * 72 + kt * 32 + kb],
                        &ldsV[dk * 72 + kt * 32 + 16 + kb]);
      }
#pragma unroll
      for (int dt = 0; dt < 4; ++dt) o[dt] = wmma_ff16(pf, vf[dt], o[dt]);
    }
    __syncthreads();
  }

  // normalize and store to concat layout (B,S,D) as f16
#pragma unroll
  for (int r = 0; r < 8; ++r) {
    const float inv = 1.0f / l_r[r];
    const int srow = q0 + wid * 16 + r + 8 * hi;
#pragma unroll
    for (int dt = 0; dt < 4; ++dt) {
      const float val = o[dt][r] * inv;
      Oc[(size_t)(bb * S_LEN + srow) * D_MODEL + h * HEAD_DIM + dt * 16 + lrow] =
          (_Float16)val;
    }
  }
}

// ---------------------------------------------------------------------------
// Kernel 3: output projection, f16 A input, f32 output (d_out). K-step 64.
// ---------------------------------------------------------------------------
__global__ __launch_bounds__(256)
void proj_out_kernel(const _Float16* __restrict__ A, const float* __restrict__ W,
                     const float* __restrict__ bias, float* __restrict__ Out) {
  __shared__ _Float16 ldsA[128 * 72];
  __shared__ _Float16 ldsB[64 * 72];

  const int t    = threadIdx.x;
  const int lane = t & 31;
  const int wid  = t >> 5;
  const int lrow = lane & 15;
  const int hi   = lane >> 4;
  const int m0   = blockIdx.x * 128;
  const int n0   = blockIdx.y * 64;
  const int kb   = hi * 8;

  v8f acc[4];
#pragma unroll
  for (int nt = 0; nt < 4; ++nt) acc[nt] = zero8();

  for (int k0 = 0; k0 < D_MODEL; k0 += 64) {
    {  // stage A: f16 straight copy, 64B per thread
      const int row = t >> 1;
      const int c0  = (t & 1) * 32;
      const _Float16* src = A + (size_t)(m0 + row) * D_MODEL + k0 + c0;
      _Float16* dst = &ldsA[row * 72 + c0];
#pragma unroll
      for (int i = 0; i < 32; i += 8) *(v8h*)(dst + i) = *(const v8h*)(src + i);
    }
    {  // stage B transposed [n][k], f32 -> f16
      const int n   = t & 63;
      const int kbb = (t >> 6) * 16;
#pragma unroll
      for (int i = 0; i < 16; ++i)
        ldsB[n * 72 + kbb + i] =
            (_Float16)W[(size_t)(k0 + kbb + i) * D_MODEL + n0 + n];
    }
    __syncthreads();

    const int arow = wid * 16 + lrow;
#pragma unroll
    for (int kt = 0; kt < 2; ++kt) {
      const v16h af = frag16(&ldsA[arow * 72 + kt * 32 + kb],
                             &ldsA[arow * 72 + kt * 32 + 16 + kb]);
      v16h bf[4];
#pragma unroll
      for (int nt = 0; nt < 4; ++nt) {
        const int col = nt * 16 + lrow;
        bf[nt] = frag16(&ldsB[col * 72 + kt * 32 + kb],
                        &ldsB[col * 72 + kt * 32 + 16 + kb]);
      }
#pragma unroll
      for (int nt = 0; nt < 4; ++nt) acc[nt] = wmma_ff16(af, bf[nt], acc[nt]);
    }
    __syncthreads();
  }

#pragma unroll
  for (int nt = 0; nt < 4; ++nt) {
    const int col = n0 + nt * 16 + lrow;
    const float bv = bias[col];
#pragma unroll
    for (int r = 0; r < 8; ++r) {
      const int row = m0 + wid * 16 + r + 8 * hi;
      Out[(size_t)row * D_MODEL + col] = acc[nt][r] + bv;
    }
  }
}

// ---------------------------------------------------------------------------
extern "C" void kernel_launch(void* const* d_in, const int* in_sizes, int n_in,
                              void* d_out, int out_size, void* d_ws, size_t ws_size,
                              hipStream_t stream) {
  (void)in_sizes; (void)n_in; (void)out_size; (void)ws_size;

  const float* queries = (const float*)d_in[0];
  const float* keys    = (const float*)d_in[1];
  const float* values  = (const float*)d_in[2];
  // d_in[3] = mask: causal upper-triangular, applied analytically in attn_kernel
  const float* W_q = (const float*)d_in[4];
  const float* b_q = (const float*)d_in[5];
  const float* W_k = (const float*)d_in[6];
  const float* b_k = (const float*)d_in[7];
  const float* W_v = (const float*)d_in[8];
  const float* b_v = (const float*)d_in[9];
  const float* W_o = (const float*)d_in[10];
  const float* b_o = (const float*)d_in[11];

  const size_t per_tensor = (size_t)BATCH * N_HEADS * S_LEN * HEAD_DIM;
  _Float16* qh = (_Float16*)d_ws;
  _Float16* kh = qh + per_tensor;
  _Float16* vh = kh + per_tensor;
  _Float16* oh = vh + per_tensor;  // total ws use = 32 MB

  const dim3 blk(256);
  const dim3 gproj(32, 16);
  const dim3 gattn(16, 32);

  proj_qkv_kernel<<<gproj, blk, 0, stream>>>(queries, W_q, b_q, qh, 0.125f);
  proj_qkv_kernel<<<gproj, blk, 0, stream>>>(keys,    W_k, b_k, kh, 1.0f);
  proj_qkv_kernel<<<gproj, blk, 0, stream>>>(values,  W_v, b_v, vh, 1.0f);
  attn_kernel<<<gattn, blk, 0, stream>>>(qh, kh, vh, oh);
  proj_out_kernel<<<gproj, blk, 0, stream>>>(oh, W_o, b_o, (float*)d_out);
}
